// JITMetalsLSTMBlock_3006477107707
// MI455X (gfx1250) — matrix operand order repaired and checked by reading, not compiled, not run on hardware
//
#include <hip/hip_runtime.h>
#include <hip/hip_bf16.h>
#include <math.h>

typedef unsigned short u16;
typedef __attribute__((ext_vector_type(16))) __bf16 v16bf;
typedef __attribute__((ext_vector_type(8)))  float  v8f;

union FragB16 { v16bf v; uint4 q[2]; };

#define B_    8
#define S_    2048
#define D_    1024
#define P_    1365
#define M_    (B_*S_)      // 16384 tokens
#define P3_   (3*P_)       // 4095 gate rows
#define NGC_  (4*P_)       // 5460 combined gate+cell rows
#define NGCP_ 5504         // padded to multiple of 64
#define PPAD_ 1376         // P padded to multiple of 32

// ---------- bf16 helpers (RNE) ----------
__device__ __forceinline__ u16 f2bf(float f) {
  union { float f; unsigned u; } a; a.f = f;
  unsigned r = (a.u + 0x7FFFu + ((a.u >> 16) & 1u)) >> 16;
  return (u16)r;
}
__device__ __forceinline__ float bf2f(u16 b) {
  union { unsigned u; float f; } a; a.u = ((unsigned)b) << 16;
  return a.f;
}

// ---------- weight conversion ----------
__global__ void __launch_bounds__(256) cvt_wgc_kernel(const float* __restrict__ gw,
                                                      const float* __restrict__ cw,
                                                      u16* __restrict__ Wgc) {
  size_t idx = (size_t)blockIdx.x * 256 + threadIdx.x;   // over NGCP_*D_
  int row = (int)(idx >> 10);
  int col = (int)(idx & 1023);
  float v = 0.f;
  if (row < P3_)       v = gw[(size_t)row * D_ + col];
  else if (row < NGC_) v = cw[(size_t)(row - P3_) * D_ + col];
  Wgc[idx] = f2bf(v);
}

__global__ void __launch_bounds__(256) cvt_wout_kernel(const float* __restrict__ ow,
                                                       u16* __restrict__ Wout) {
  size_t idx = (size_t)blockIdx.x * 256 + threadIdx.x;   // over D_*PPAD_
  int d = (int)(idx / PPAD_);
  int p = (int)(idx % PPAD_);
  float v = (p < P_) ? ow[(size_t)d * P_ + p] : 0.f;
  Wout[idx] = f2bf(v);
}

// ---------- RMSNorm: one block per token row ----------
__global__ void __launch_bounds__(256) rmsnorm_kernel(const float* __restrict__ x,
                                                      const float* __restrict__ w,
                                                      u16* __restrict__ xn) {
  int row = blockIdx.x;
  int tid = threadIdx.x;
  const float4* xr = (const float4*)(x + (size_t)row * D_);
  float4 v = xr[tid];
  float ss = v.x * v.x + v.y * v.y + v.z * v.z + v.w * v.w;
  #pragma unroll
  for (int off = 16; off; off >>= 1) ss += __shfl_down(ss, off, 32);
  __shared__ float part[8];
  if ((tid & 31) == 0) part[tid >> 5] = ss;
  __syncthreads();
  float tot = 0.f;
  #pragma unroll
  for (int i = 0; i < 8; ++i) tot += part[i];
  float rs = rsqrtf(tot * (1.0f / (float)D_) + 1e-6f);
  float4 wv = ((const float4*)w)[tid];
  u16* dst = xn + (size_t)row * D_ + tid * 4;
  dst[0] = f2bf(v.x * rs * wv.x);
  dst[1] = f2bf(v.y * rs * wv.y);
  dst[2] = f2bf(v.z * rs * wv.z);
  dst[3] = f2bf(v.w * rs * wv.w);
}

// ---------- fused gates+cell GEMM: one wave per 64x64 tile ----------
__device__ __forceinline__ void store_gate(u16* __restrict__ iP, u16* __restrict__ fP,
                                           u16* __restrict__ oP, u16* __restrict__ cP,
                                           int m, int n, float v) {
  if (n < P3_) {
    float t = 15.0f * tanhf(v * (1.0f / 15.0f));
    float g = 1.0f / (1.0f + __expf(-t));
    int plane = n / P_;
    int p = n - plane * P_;
    u16* dst = (plane == 0) ? iP : ((plane == 1) ? fP : oP);
    dst[(size_t)m * PPAD_ + p] = f2bf(g);
  } else if (n < NGC_) {
    cP[(size_t)m * PPAD_ + (n - P3_)] = f2bf(v);
  }
}

__global__ void __launch_bounds__(32) gemm_gates_kernel(const u16* __restrict__ Xn,
                                                        const u16* __restrict__ Wgc,
                                                        u16* __restrict__ iP, u16* __restrict__ fP,
                                                        u16* __restrict__ oP, u16* __restrict__ cP) {
  int m0 = blockIdx.x * 64;
  int n0 = blockIdx.y * 64;
  int lane = threadIdx.x & 31;
  int lh  = lane >> 4;     // lane half
  int l15 = lane & 15;

  // A fragment lane addressing (16-bit A 16x32): two 16B chunks at K {0..7,16..23} / {8..15,24..31}
  // B fragment lane addressing (16-bit B 32x16): two 16B chunks at K {0..7,16..23} / {8..15,24..31}
  const u16* ap[4];
  const u16* bp[4];
  #pragma unroll
  for (int i = 0; i < 4; ++i) {
    ap[i] = Xn  + (size_t)(m0 + i * 16 + l15) * D_ + (lh ? 8 : 0);
    bp[i] = Wgc + (size_t)(n0 + i * 16 + l15) * D_ + (lh ? 8 : 0);
  }

  v8f acc[4][4];
  #pragma unroll
  for (int im = 0; im < 4; ++im)
    #pragma unroll
    for (int in2 = 0; in2 < 4; ++in2)
      acc[im][in2] = (v8f){};

  for (int k = 0; k < D_; k += 32) {
    FragB16 a[4], b[4];
    #pragma unroll
    for (int i = 0; i < 4; ++i) {
      a[i].q[0] = *(const uint4*)(ap[i] + k);
      a[i].q[1] = *(const uint4*)(ap[i] + k + 16);
      b[i].q[0] = *(const uint4*)(bp[i] + k);
      b[i].q[1] = *(const uint4*)(bp[i] + k + 16);
    }
    #pragma unroll
    for (int im = 0; im < 4; ++im)
      #pragma unroll
      for (int in2 = 0; in2 < 4; ++in2)
        acc[im][in2] = __builtin_amdgcn_wmma_f32_16x16x32_bf16(
            false, a[im].v, false, b[in2].v, (short)0, acc[im][in2], false, false);
  }

  #pragma unroll
  for (int im = 0; im < 4; ++im) {
    #pragma unroll
    for (int in2 = 0; in2 < 4; ++in2) {
      v8f c = acc[im][in2];
      int n = n0 + in2 * 16 + l15;
      #pragma unroll
      for (int r = 0; r < 8; ++r) {
        int m = m0 + im * 16 + r + (lh ? 8 : 0);
        store_gate(iP, fP, oP, cP, m, n, c[r]);
      }
    }
  }
}

// ---------- sequential LSTM-style scan: one thread per (b,p) ----------
__global__ void __launch_bounds__(256) scan_kernel(const u16* __restrict__ iP,
                                                   const u16* __restrict__ fP,
                                                   const u16* __restrict__ oP,
                                                   const u16* __restrict__ cP,
                                                   const float* __restrict__ h0,
                                                   u16* __restrict__ Y,
                                                   float* __restrict__ hfin) {
  int p = blockIdx.x * 256 + threadIdx.x;
  int b = blockIdx.y;
  if (p >= PPAD_) return;
  size_t base = (size_t)b * S_ * PPAD_ + p;
  if (p >= P_) {                     // zero the K padding of Y for the out-GEMM
    for (int t = 0; t < S_; ++t) Y[base + (size_t)t * PPAD_] = 0;
    return;
  }
  float h = h0[(size_t)b * P_ + p];
  for (int t = 0; t < S_; ++t) {
    size_t idx = base + (size_t)t * PPAD_;
    float ct = bf2f(cP[idx]);
    float it = bf2f(iP[idx]);
    float ft = bf2f(fP[idx]);
    float ot = bf2f(oP[idx]);
    h = ft * h + it * tanhf(ct);
    Y[idx] = f2bf(ot * tanhf(h));
  }
  hfin[(size_t)b * P_ + p] = h;
}

// ---------- output GEMM with residual: one wave per 64x64 tile ----------
__global__ void __launch_bounds__(32) gemm_out_kernel(const u16* __restrict__ Y,
                                                      const u16* __restrict__ Wout,
                                                      const float* __restrict__ x,
                                                      float* __restrict__ out) {
  int m0 = blockIdx.x * 64;
  int n0 = blockIdx.y * 64;
  int lane = threadIdx.x & 31;
  int lh  = lane >> 4;
  int l15 = lane & 15;

  const u16* ap[4];
  const u16* bp[4];
  #pragma unroll
  for (int i = 0; i < 4; ++i) {
    ap[i] = Y    + (size_t)(m0 + i * 16 + l15) * PPAD_ + (lh ? 8 : 0);
    bp[i] = Wout + (size_t)(n0 + i * 16 + l15) * PPAD_ + (lh ? 8 : 0);
  }

  v8f acc[4][4];
  #pragma unroll
  for (int im = 0; im < 4; ++im)
    #pragma unroll
    for (int in2 = 0; in2 < 4; ++in2)
      acc[im][in2] = (v8f){};

  for (int k = 0; k < PPAD_; k += 32) {
    FragB16 a[4], b[4];
    #pragma unroll
    for (int i = 0; i < 4; ++i) {
      a[i].q[0] = *(const uint4*)(ap[i] + k);
      a[i].q[1] = *(const uint4*)(ap[i] + k + 16);
      b[i].q[0] = *(const uint4*)(bp[i] + k);
      b[i].q[1] = *(const uint4*)(bp[i] + k + 16);
    }
    #pragma unroll
    for (int im = 0; im < 4; ++im)
      #pragma unroll
      for (int in2 = 0; in2 < 4; ++in2)
        acc[im][in2] = __builtin_amdgcn_wmma_f32_16x16x32_bf16(
            false, a[im].v, false, b[in2].v, (short)0, acc[im][in2], false, false);
  }

  #pragma unroll
  for (int im = 0; im < 4; ++im) {
    #pragma unroll
    for (int in2 = 0; in2 < 4; ++in2) {
      v8f c = acc[im][in2];
      int n = n0 + in2 * 16 + l15;
      #pragma unroll
      for (int r = 0; r < 8; ++r) {
        int m = m0 + im * 16 + r + (lh ? 8 : 0);
        size_t o = (size_t)m * D_ + n;
        out[o] = x[o] + c[r];
      }
    }
  }
}

extern "C" void kernel_launch(void* const* d_in, const int* in_sizes, int n_in,
                              void* d_out, int out_size, void* d_ws, size_t ws_size,
                              hipStream_t stream) {
  (void)in_sizes; (void)n_in; (void)out_size; (void)ws_size;
  const float* x      = (const float*)d_in[0];
  const float* h0     = (const float*)d_in[1];
  const float* lnw    = (const float*)d_in[2];
  const float* gate_w = (const float*)d_in[3];
  const float* cell_w = (const float*)d_in[4];
  const float* out_w  = (const float*)d_in[5];
  float* outp = (float*)d_out;
  float* hfin = outp + (size_t)M_ * D_;

  char* w = (char*)d_ws;
  u16* Xn   = (u16*)w;  w += (size_t)M_ * D_ * 2;
  u16* Wgc  = (u16*)w;  w += (size_t)NGCP_ * D_ * 2;
  u16* Wout = (u16*)w;  w += (size_t)D_ * PPAD_ * 2;
  u16* iP   = (u16*)w;  w += (size_t)M_ * PPAD_ * 2;
  u16* fP   = (u16*)w;  w += (size_t)M_ * PPAD_ * 2;
  u16* oP   = (u16*)w;  w += (size_t)M_ * PPAD_ * 2;
  u16* cP   = (u16*)w;  w += (size_t)M_ * PPAD_ * 2;
  u16* Y    = (u16*)w;  w += (size_t)M_ * PPAD_ * 2;

  cvt_wgc_kernel<<<(NGCP_ * D_) / 256, 256, 0, stream>>>(gate_w, cell_w, Wgc);
  cvt_wout_kernel<<<(D_ * PPAD_) / 256, 256, 0, stream>>>(out_w, Wout);
  rmsnorm_kernel<<<M_, 256, 0, stream>>>(x, lnw, Xn);
  gemm_gates_kernel<<<dim3(M_ / 64, NGCP_ / 64), 32, 0, stream>>>(Xn, Wgc, iP, fP, oP, cP);
  scan_kernel<<<dim3((PPAD_ + 255) / 256, B_), 256, 0, stream>>>(iP, fP, oP, cP, h0, Y, hfin);
  gemm_out_kernel<<<dim3(M_ / 64, D_ / 64), 32, 0, stream>>>(Y, Wout, x, outp);
}
